// HawkResidualBlock_16904991277202
// MI455X (gfx1250) — matrix operand order repaired and checked by reading, not compiled, not run on hardware
//
#include <hip/hip_runtime.h>
#include <hip/hip_bf16.h>
#include <stdint.h>

typedef __bf16 bh;
typedef __attribute__((ext_vector_type(16))) __bf16 v16bf;
typedef __attribute__((ext_vector_type(8)))  __bf16 v8bf;
typedef __attribute__((ext_vector_type(8)))  float  v8f;
typedef __attribute__((ext_vector_type(4)))  int    v4i;
typedef __attribute__((ext_vector_type(4)))  unsigned int v4ui;

#define B_  2
#define S_  2048
#define D_  2048
#define H_  16
#define HD_ 128
#define F_  8192
#define KCONV_ 4
#define ROWS_ (B_*S_)                       // 4096 tokens
static const long T_  = (long)B_*S_*D_;     // 8388608
static const long FF_ = (long)B_*S_*F_;     // 33554432

// ---- gfx1250 async global->LDS staging (ASYNCcnt path) -------------------
#if defined(__has_builtin)
#  if __has_builtin(__builtin_amdgcn_global_load_async_to_lds_b128)
#    define ASYNC_LDS 1
#  endif
#endif
#ifndef ASYNC_LDS
#  define ASYNC_LDS 0
#endif

typedef __attribute__((address_space(1))) v4i* gptr_v4i;
typedef __attribute__((address_space(3))) v4i* lptr_v4i;

__device__ __forceinline__ void stage16(const bh* __restrict__ g, bh* l) {
#if ASYNC_LDS
  __builtin_amdgcn_global_load_async_to_lds_b128(
      (gptr_v4i)(const_cast<bh*>(g)), (lptr_v4i)l, 0, 0);
#else
  *(v8bf*)l = *(const v8bf*)g;
#endif
}

__device__ __forceinline__ void async_drain() {
#if ASYNC_LDS
#  if __has_builtin(__builtin_amdgcn_s_wait_asynccnt)
  __builtin_amdgcn_s_wait_asynccnt(0);
#  else
  asm volatile("s_wait_asynccnt 0" ::: "memory");
#  endif
#endif
}

// ---------------------------------------------------------------------------
// WMMA bf16 GEMM:  C[M,N] = A[M,K]*B[K,N] (+bias[n]), row-major, batched.
// BM=128, BN=128, BK=32, double-buffered LDS (ping-pong). 256 threads =
// 8 wave32 waves (2x4); each wave owns a 64x32 tile -> 4x2
// v_wmma_f32_16x16x32_bf16 per K-step. All dims divide the tiles so EXEC
// stays all-ones (WMMA requirement). Per iteration: issue next tile's async-A
// (global->LDS DMA) + B b128 global loads, run 8 WMMAs on the current tile,
// then v_perm-pack + ds_store_b32 the B tile transposed, drain, barrier.
// ---------------------------------------------------------------------------
#define BM 128
#define BN 128
#define BK 32
#define LDP 40   // BK + 8 bf16 pad -> 80B row stride, keeps 16B alignment

__global__ __launch_bounds__(256)
void wmma_gemm_bf16(const bh* __restrict__ A, const bh* __restrict__ Bw,
                    float* __restrict__ C, const float* __restrict__ bias,
                    int M, int N, int K, int lda, int ldb, int ldc,
                    long sA, long sB, long sC, int sBias)
{
  __shared__ bh As[2 * BM * LDP];
  __shared__ bh Bs[2 * BN * LDP];   // stored transposed: [n][k]

  const int z = blockIdx.z;
  A  += (long)z * sA;
  Bw += (long)z * sB;
  C  += (long)z * sC;

  const int tid  = threadIdx.x;        // 0..255
  const int lane = tid & 31;
  const int wave = tid >> 5;           // 0..7
  const int wr   = (wave >> 2) * 64;   // wave row offset within 128
  const int wc   = (wave & 3) * 32;    // wave col offset within 128
  const int half = lane >> 4;          // 0|1 (K-half for A/B, M-half for C)
  const int l16  = lane & 15;
  const long row0 = (long)blockIdx.x * BM;
  const int  col0 = blockIdx.y * BN;

  // ---- staging addresses hoisted out of the K loop (pointer bumping only) --
  // A tile 128x32: 512 x 16B chunks; thread handles chunks tid and tid+256.
  const int ar0 = tid >> 2,          ac0 = (tid & 3) * 8;
  const int ar1 = (tid + 256) >> 2,  ac1 = ((tid + 256) & 3) * 8;
  const bh* aSrc0 = A + (row0 + ar0) * (long)lda + ac0;
  const bh* aSrc1 = A + (row0 + ar1) * (long)lda + ac1;
  const int aoff0 = ar0 * LDP + ac0;
  const int aoff1 = ar1 * LDP + ac1;
  // B tile 32x128: thread owns K rows {bkp, bkp+1} x cols [bn0, bn0+8).
  // Global side: two coalesced b128 loads; LDS side: 8 v_perm packs + b32
  // stores into the transposed layout Bs[n][k] (k pair contiguous, 4B aligned).
  const int bkp = (tid >> 4) * 2;      // 0,2,...,30
  const int bn0 = (tid & 15) * 8;      // 0,8,...,120
  const bh* bSrc = Bw + (long)bkp * ldb + col0 + bn0;
  const long ldbBK = (long)BK * ldb;
  const int boff = bn0 * LDP + bkp;

  v8f acc[4][2] = {};
  const int nIter = K / BK;

  // ---- prologue: fill buffer 0 ----
  stage16(aSrc0, &As[aoff0]);
  stage16(aSrc1, &As[aoff1]);
  {
    v4ui r0 = *(const v4ui*)bSrc;          // row bkp   (8 bf16)
    v4ui r1 = *(const v4ui*)(bSrc + ldb);  // row bkp+1 (8 bf16)
#pragma unroll
    for (int d2 = 0; d2 < 4; ++d2) {
      uint32_t lo = __builtin_amdgcn_perm(r1[d2], r0[d2], 0x05040100u);
      uint32_t hi = __builtin_amdgcn_perm(r1[d2], r0[d2], 0x07060302u);
      *(uint32_t*)&Bs[boff + (2 * d2) * LDP]     = lo;
      *(uint32_t*)&Bs[boff + (2 * d2 + 1) * LDP] = hi;
    }
  }
  aSrc0 += BK; aSrc1 += BK; bSrc += ldbBK;
  async_drain();
  __syncthreads();

  for (int it = 0; it < nIter; ++it) {
    const int cur = it & 1;
    const bh* as = As + cur * (BM * LDP);
    const bh* bs = Bs + cur * (BN * LDP);
    bh* asN = As + (cur ^ 1) * (BM * LDP);
    bh* bsN = Bs + (cur ^ 1) * (BN * LDP);
    const bool hasNext = (it + 1 < nIter);   // uniform

    // ---- issue next tile's loads before computing this one ----
    v4ui r0, r1;
    if (hasNext) {
      stage16(aSrc0, asN + aoff0);           // async global->LDS DMA
      stage16(aSrc1, asN + aoff1);
      r0 = *(const v4ui*)bSrc;               // b128, held in regs over compute
      r1 = *(const v4ui*)(bSrc + ldb);
      aSrc0 += BK; aSrc1 += BK; bSrc += ldbBK;
      __builtin_prefetch(aSrc0, 0, 0);
      __builtin_prefetch(bSrc, 0, 0);
    }

    // ---- fragments per ISA 16-bit layouts ----
    v16bf af[4], bfr[2];
#pragma unroll
    for (int i = 0; i < 4; ++i) {      // A 16x32: lane holds m=l16, k=half*8+e / +16
      int m  = wr + i * 16 + l16;
      int kb = half * 8;
      v8bf lo = *(const v8bf*)&as[m * LDP + kb];
      v8bf hi = *(const v8bf*)&as[m * LDP + kb + 16];
      af[i] = __builtin_shufflevector(lo, hi, 0,1,2,3,4,5,6,7,8,9,10,11,12,13,14,15);
    }
#pragma unroll
    for (int j = 0; j < 2; ++j) {      // B 32x16: lane holds n=l16, k=half*8+e / +16
      int n  = wc + j * 16 + l16;
      int kb = half * 8;
      v8bf lo = *(const v8bf*)&bs[n * LDP + kb];
      v8bf hi = *(const v8bf*)&bs[n * LDP + kb + 16];
      bfr[j] = __builtin_shufflevector(lo, hi, 0,1,2,3,4,5,6,7,8,9,10,11,12,13,14,15);
    }
#pragma unroll
    for (int i = 0; i < 4; ++i)
#pragma unroll
      for (int j = 0; j < 2; ++j)
        acc[i][j] = __builtin_amdgcn_wmma_f32_16x16x32_bf16(
            false, af[i], false, bfr[j], (short)0, acc[i][j], false, false);

    // ---- transpose-pack next B tile into LDS after compute ----
    if (hasNext) {
#pragma unroll
      for (int d2 = 0; d2 < 4; ++d2) {
        uint32_t lo = __builtin_amdgcn_perm(r1[d2], r0[d2], 0x05040100u);
        uint32_t hi = __builtin_amdgcn_perm(r1[d2], r0[d2], 0x07060302u);
        *(uint32_t*)&bsN[boff + (2 * d2) * LDP]     = lo;
        *(uint32_t*)&bsN[boff + (2 * d2 + 1) * LDP] = hi;
      }
    }
    async_drain();
    __syncthreads();
  }

  // ---- epilogue: C layout lane holds n=l16, m=half*8+v ----
#pragma unroll
  for (int j = 0; j < 2; ++j) {
    int nn = col0 + wc + j * 16 + l16;
    float bv = bias ? bias[(long)z * sBias + nn] : 0.f;
#pragma unroll
    for (int i = 0; i < 4; ++i) {
      long mbase = row0 + wr + i * 16 + half * 8;
#pragma unroll
      for (int v = 0; v < 8; ++v)
        C[(mbase + v) * (long)ldc + nn] = acc[i][j][v] + bv;
    }
  }
}

// ---------------------------------------------------------------------------
// Elementwise / reduction helpers
// ---------------------------------------------------------------------------
__global__ void f32_to_bf16_kernel(const float* __restrict__ in,
                                   bh* __restrict__ out, long n) {
  long i = (long)blockIdx.x * blockDim.x + threadIdx.x;
  if (i < n) out[i] = (bh)in[i];
}

__global__ __launch_bounds__(256)
void rmsnorm_bf16_kernel(const float* __restrict__ x, const float* __restrict__ w,
                         bh* __restrict__ out) {
  const long row = blockIdx.x;
  const float* xr = x + row * D_;
  float ss = 0.f;
  for (int d = threadIdx.x; d < D_; d += 256) { float v = xr[d]; ss += v * v; }
#pragma unroll
  for (int o = 16; o > 0; o >>= 1) ss += __shfl_down(ss, o, 32);
  __shared__ float red[8];
  __shared__ float s_sc;
  if ((threadIdx.x & 31) == 0) red[threadIdx.x >> 5] = ss;
  __syncthreads();
  if (threadIdx.x == 0) {
    float t = 0.f;
    for (int i = 0; i < 8; ++i) t += red[i];
    s_sc = rsqrtf(t / (float)D_ + 1e-6f);
  }
  __syncthreads();
  float sc = s_sc;
  bh* orow = out + row * D_;
  for (int d = threadIdx.x; d < D_; d += 256) orow[d] = (bh)(xr[d] * sc * w[d]);
}

__device__ __forceinline__ float gelu_tanh(float x) {
  float x3 = x * x * x;
  return 0.5f * x * (1.f + tanhf(0.7978845608028654f * (x + 0.044715f * x3)));
}

__global__ void gelu_kernel(float* __restrict__ y, long n) {
  long i = (long)blockIdx.x * blockDim.x + threadIdx.x;
  if (i < n) y[i] = gelu_tanh(y[i]);
}

__global__ void conv_kernel(const float* __restrict__ xb, const float* __restrict__ cw,
                            const float* __restrict__ cb, float* __restrict__ xc,
                            bh* __restrict__ xcbf) {
  long i = (long)blockIdx.x * blockDim.x + threadIdx.x;
  if (i >= T_) return;
  int d = (int)(i % D_);
  long bs = i / D_;
  int s = (int)(bs % S_);
  float acc = cb[d];
#pragma unroll
  for (int k = 0; k < KCONV_; ++k) {
    int sp = s + k - (KCONV_ - 1);
    if (sp >= 0) acc += cw[k * D_ + d] * xb[i + (long)(k - (KCONV_ - 1)) * D_];
  }
  xc[i] = acc;
  xcbf[i] = (bh)acc;
}

// rg: agate logits -> a ; ig: igate logits -> bx = mult*i*xc   (in place)
__global__ void gatefuse_kernel(float* __restrict__ rg, float* __restrict__ ig,
                                const float* __restrict__ xc,
                                const float* __restrict__ ap) {
  long i = (long)blockIdx.x * blockDim.x + threadIdx.x;
  if (i >= T_) return;
  int d = (int)(i % D_);
  float r  = 1.f / (1.f + expf(-rg[i]));
  float ii = 1.f / (1.f + expf(-ig[i]));
  float z = -ap[d];
  float sp = (z > 0.f) ? (z + log1pf(expf(-z))) : log1pf(expf(z));  // softplus(-a_param)
  float la = -8.f * r * sp;                                         // C_RGLRU = 8
  float a  = expf(la);
  float mult = sqrtf(fmaxf(1.f - expf(2.f * la), 0.f));
  rg[i] = a;
  ig[i] = mult * ii * xc[i];
}

// RG-LRU scan: parallel over B*D channels, sequential over S; coalesced steps.
__global__ __launch_bounds__(256)
void scan_kernel(const float* __restrict__ a, const float* __restrict__ bx,
                 const float* __restrict__ h0, float* __restrict__ lru,
                 float* __restrict__ hlast) {
  int ch = blockIdx.x * 256 + threadIdx.x;   // 0..B*D-1
  int b = ch / D_;
  int d = ch % D_;
  float h = h0[ch];
  long base = (long)b * S_ * D_ + d;
  for (int s = 0; s < S_; ++s) {
    long p = base + (long)s * D_;
    h = fmaf(a[p], h, bx[p]);
    lru[p] = h;
  }
  hlast[ch] = h;
}

__global__ void mul_bf16_kernel(const float* __restrict__ a, const float* __restrict__ b,
                                bh* __restrict__ o, long n) {
  long i = (long)blockIdx.x * blockDim.x + threadIdx.x;
  if (i < n) o[i] = (bh)(a[i] * b[i]);
}

__global__ void add_kernel(const float* __restrict__ a, const float* __restrict__ b,
                           float* __restrict__ o, long n) {
  long i = (long)blockIdx.x * blockDim.x + threadIdx.x;
  if (i < n) o[i] = a[i] + b[i];
}

__global__ void gelumul_bf16_kernel(const float* __restrict__ g, const float* __restrict__ u,
                                    bh* __restrict__ o, long n) {
  long i = (long)blockIdx.x * blockDim.x + threadIdx.x;
  if (i < n) o[i] = (bh)(gelu_tanh(g[i]) * u[i]);
}

// ---------------------------------------------------------------------------
extern "C" void kernel_launch(void* const* d_in, const int* in_sizes, int n_in,
                              void* d_out, int out_size, void* d_ws, size_t ws_size,
                              hipStream_t stream) {
  (void)in_sizes; (void)n_in; (void)out_size; (void)ws_size;
  const float* x    = (const float*)d_in[0];
  const float* h0   = (const float*)d_in[1];
  const float* r1w  = (const float*)d_in[2];
  const float* r2w  = (const float*)d_in[3];
  const float* Wx   = (const float*)d_in[4];
  const float* Wy   = (const float*)d_in[5];
  const float* cw   = (const float*)d_in[6];
  const float* cb   = (const float*)d_in[7];
  const float* ap   = (const float*)d_in[8];
  const float* igw  = (const float*)d_in[9];
  const float* igb  = (const float*)d_in[10];
  const float* agw  = (const float*)d_in[11];
  const float* agb  = (const float*)d_in[12];
  const float* Wout = (const float*)d_in[13];
  const float* Wg   = (const float*)d_in[14];
  const float* Wu   = (const float*)d_in[15];
  const float* Wd   = (const float*)d_in[16];
  float* out = (float*)d_out;                 // [T_] out, then [B*D] h_last

  char* ws = (char*)d_ws;
  size_t off = 0;
  auto alloc = [&](size_t bytes) -> void* {
    void* p = ws + off;
    off = (off + bytes + 255) & ~(size_t)255;
    return p;
  };
  const size_t DD = (size_t)D_ * D_, DF = (size_t)D_ * F_, GW = (size_t)H_ * HD_ * HD_;
  bh* wx_bf  = (bh*)alloc(DD * 2);
  bh* wy_bf  = (bh*)alloc(DD * 2);
  bh* wo_bf  = (bh*)alloc(DD * 2);
  bh* wg_bf  = (bh*)alloc(DF * 2);
  bh* wu_bf  = (bh*)alloc(DF * 2);
  bh* wd_bf  = (bh*)alloc(DF * 2);
  bh* igw_bf = (bh*)alloc(GW * 2);
  bh* agw_bf = (bh*)alloc(GW * 2);
  bh* abf    = (bh*)alloc((size_t)T_ * 2);    // rotating bf16 A-operand
  bh* gu_bf  = (bh*)alloc((size_t)FF_ * 2);
  float* ybuf  = (float*)alloc((size_t)T_ * 4);
  float* xbbuf = (float*)alloc((size_t)T_ * 4);
  float* xcbuf = (float*)alloc((size_t)T_ * 4);
  float* rgbuf = (float*)alloc((size_t)T_ * 4);
  float* igbuf = (float*)alloc((size_t)T_ * 4);
  float* resid = (float*)alloc((size_t)T_ * 4);
  float* gbuf  = (float*)alloc((size_t)FF_ * 4);
  float* ubuf  = (float*)alloc((size_t)FF_ * 4);

  auto cvt = [&](const float* src, bh* dst, long n) {
    f32_to_bf16_kernel<<<dim3((unsigned)((n + 255) / 256)), dim3(256), 0, stream>>>(src, dst, n);
  };
  cvt(Wx, wx_bf, (long)DD);   cvt(Wy, wy_bf, (long)DD);   cvt(Wout, wo_bf, (long)DD);
  cvt(Wg, wg_bf, (long)DF);   cvt(Wu, wu_bf, (long)DF);   cvt(Wd, wd_bf, (long)DF);
  cvt(igw, igw_bf, (long)GW); cvt(agw, agw_bf, (long)GW);

  auto gemm = [&](const bh* A, const bh* Bm, float* C, const float* bias,
                  int M, int N, int K, int lda, int ldb, int ldc,
                  long sA, long sB, long sC, int sBias, int batch) {
    dim3 g(M / BM, N / BN, batch);
    wmma_gemm_bf16<<<g, dim3(256), 0, stream>>>(A, Bm, C, bias, M, N, K,
                                                lda, ldb, ldc, sA, sB, sC, sBias);
  };
  const unsigned gT  = (unsigned)((T_ + 255) / 256);
  const unsigned gFF = (unsigned)((FF_ + 255) / 256);

  // 1) temporal pre-norm -> bf16
  rmsnorm_bf16_kernel<<<ROWS_, 256, 0, stream>>>(x, r1w, abf);
  // 2) y = normed@Wy ; xb = normed@Wx
  gemm(abf, wy_bf, ybuf,  nullptr, ROWS_, D_, D_, D_, D_, D_, 0, 0, 0, 0, 1);
  gemm(abf, wx_bf, xbbuf, nullptr, ROWS_, D_, D_, D_, D_, D_, 0, 0, 0, 0, 1);
  // 3) gelu gate branch
  gelu_kernel<<<gT, 256, 0, stream>>>(ybuf, T_);
  // 4) causal depthwise conv (k=4) -> xc (f32) + bf16 copy for gate GEMMs
  conv_kernel<<<gT, 256, 0, stream>>>(xbbuf, cw, cb, xcbuf, abf);
  // 5) per-head gate projections (batched over H=16 heads) + bias
  gemm(abf, agw_bf, rgbuf, agb, ROWS_, HD_, HD_, D_, HD_, D_,
       (long)HD_, (long)HD_ * HD_, (long)HD_, HD_, H_);
  gemm(abf, igw_bf, igbuf, igb, ROWS_, HD_, HD_, D_, HD_, D_,
       (long)HD_, (long)HD_ * HD_, (long)HD_, HD_, H_);
  // 6) RG-LRU gate math: rg->a, ig->bx
  gatefuse_kernel<<<gT, 256, 0, stream>>>(rgbuf, igbuf, xcbuf, ap);
  // 7) linear recurrence; lru reuses xcbuf; h_last straight into d_out tail
  scan_kernel<<<(B_ * D_) / 256, 256, 0, stream>>>(rgbuf, igbuf, h0, xcbuf, out + T_);
  // 8) (y * lru) -> bf16
  mul_bf16_kernel<<<gT, 256, 0, stream>>>(ybuf, xcbuf, abf, T_);
  // 9) temporal_out = (y*lru)@Wout  (into xbbuf)
  gemm(abf, wo_bf, xbbuf, nullptr, ROWS_, D_, D_, D_, D_, D_, 0, 0, 0, 0, 1);
  // 10) residual = temporal_out + x
  add_kernel<<<gT, 256, 0, stream>>>(xbbuf, x, resid, T_);
  // 11) second rmsnorm -> bf16
  rmsnorm_bf16_kernel<<<ROWS_, 256, 0, stream>>>(resid, r2w, abf);
  // 12) MLP up/gate
  gemm(abf, wg_bf, gbuf, nullptr, ROWS_, F_, D_, D_, F_, F_, 0, 0, 0, 0, 1);
  gemm(abf, wu_bf, ubuf, nullptr, ROWS_, F_, D_, D_, F_, F_, 0, 0, 0, 0, 1);
  // 13) gelu(g)*u -> bf16
  gelumul_bf16_kernel<<<gFF, 256, 0, stream>>>(gbuf, ubuf, gu_bf, FF_);
  // 14) down projection (into ybuf)
  gemm(gu_bf, wd_bf, ybuf, nullptr, ROWS_, D_, F_, F_, D_, D_, 0, 0, 0, 0, 1);
  // 15) out = mlp + residual
  add_kernel<<<gT, 256, 0, stream>>>(ybuf, resid, out, T_);
}
